// HyperbolicGAIN_48266842472776
// MI455X (gfx1250) — compile-verified, matrix operand order checked by reading, not run on hardware
//
#include <hip/hip_runtime.h>
#include <hip/hip_bf16.h>

#define Nn     10000
#define Ee     100000
#define DINc   256
#define DOUTc  128
#define Hc     8
#define HIDc   128
#define BN_EPSf 1e-3f
#define MAX_TANHf 0.99999f

typedef __bf16 bf16_t;
typedef __attribute__((ext_vector_type(16))) __bf16 v16bf;
typedef __attribute__((ext_vector_type(8)))  float  v8f;
typedef __attribute__((ext_vector_type(4)))  float  v4f;

// ---------- helpers ----------
__device__ __forceinline__ bf16_t f2bf(float f) {
  unsigned u = __float_as_uint(f);
  unsigned r = (u + 0x7FFFu + ((u >> 16) & 1u)) >> 16;   // RNE truncate
  unsigned short s = (unsigned short)r;
  bf16_t b; __builtin_memcpy(&b, &s, 2); return b;
}
__device__ __forceinline__ float bf2f(bf16_t b) {
  unsigned short s; __builtin_memcpy(&s, &b, 2);
  return __uint_as_float(((unsigned)s) << 16);
}
// monotone float<->uint map so atomicMax(uint) == float max (handles negatives)
__device__ __forceinline__ unsigned fmap(float f) {
  unsigned u = __float_as_uint(f);
  return (u >> 31) ? ~u : (u | 0x80000000u);
}
__device__ __forceinline__ float funmap(unsigned u) {
  return (u & 0x80000000u) ? __uint_as_float(u & 0x7FFFFFFFu) : __uint_as_float(~u);
}
// A/B fragment load per CDNA5 16-bit 16x32 layout:
// lanes 0-15 hold K=[0..7] (v0..3) and K=[16..23] (v4..7); lanes 16-31 K=[8..15],[24..31].
// caller passes p = rowbase + kstep*32 + (lane>=16)*8
__device__ __forceinline__ v16bf ldfrag(const bf16_t* p) {
  v16bf f;
#pragma unroll
  for (int i = 0; i < 8; ++i) { f[i] = p[i]; f[i + 8] = p[i + 16]; }
  return f;
}
__device__ __forceinline__ float wred(float v) {
#pragma unroll
  for (int off = 16; off > 0; off >>= 1) v += __shfl_xor(v, off, 32);
  return v;
}

// ---------- utility kernels ----------
__global__ void k_zero(float* p, int n) {
  int i = blockIdx.x * 256 + threadIdx.x;
  if (i < n) p[i] = 0.f;
}
__global__ void k_f2bf(const float* s, bf16_t* d, int n) {
  int i = blockIdx.x * 256 + threadIdx.x;
  if (i < n) d[i] = f2bf(s[i]);
}
// d[b][c][r] = bf16(s[b][r][c])  : builds B^T so a B-matrix column is a contiguous K-row
__global__ void k_tr(const float* s, bf16_t* d, int rows, int cols) {
  int b = blockIdx.y;
  int i = blockIdx.x * 256 + threadIdx.x;
  if (i < rows * cols) {
    int r = i / cols, c = i - r * cols;
    d[(size_t)b * rows * cols + (size_t)c * rows + r] = f2bf(s[(size_t)b * rows * cols + i]);
  }
}

// ---------- projections: t = x@Wt[h], out = (1+eps)*x@Ws[h], escale = expmap0 scale ----------
__global__ void __launch_bounds__(256) k_gemm_proj(
    const bf16_t* __restrict__ xb, const bf16_t* __restrict__ wtT, const bf16_t* __restrict__ wsT,
    const float* __restrict__ curv, const float* __restrict__ epsv,
    float* __restrict__ tb, float* __restrict__ outb, float* __restrict__ esc) {
  __shared__ float rowsq[16];
  const int h = blockIdx.y, row0 = blockIdx.x * 16, tid = threadIdx.x;
  const int lane = tid & 31, w = tid >> 5, lh = lane >> 4, m = lane & 15, n0 = w * 16;
  if (tid < 16) rowsq[tid] = 0.f;

  const bf16_t* arow = xb + (size_t)(row0 + m) * DINc + lh * 8;
  const bf16_t* btr  = wtT + ((size_t)h * DOUTc + n0 + m) * DINc + lh * 8;
  const bf16_t* bsr  = wsT + ((size_t)h * DOUTc + n0 + m) * DINc + lh * 8;

  v8f acc_t = {}, acc_s = {};
#pragma unroll
  for (int kk = 0; kk < DINc / 32; ++kk) {
    v16bf a  = ldfrag(arow + kk * 32);
    v16bf bt = ldfrag(btr  + kk * 32);
    v16bf bs = ldfrag(bsr  + kk * 32);
    acc_t = __builtin_amdgcn_wmma_f32_16x16x32_bf16(false, a, false, bt, (short)0, acc_t, false, false);
    acc_s = __builtin_amdgcn_wmma_f32_16x16x32_bf16(false, a, false, bs, (short)0, acc_s, false, false);
  }
  __syncthreads();                       // rowsq init visible
#pragma unroll
  for (int r = 0; r < 8; ++r) atomicAdd(&rowsq[r + lh * 8], acc_t[r] * acc_t[r]);

  const float ep1 = 1.f + epsv[0];
#pragma unroll
  for (int r = 0; r < 8; ++r) {          // C layout: m = r + 8*half, n = lane&15
    size_t o = ((size_t)h * Nn + row0 + r + lh * 8) * DOUTc + n0 + m;
    tb[o]   = acc_t[r];
    outb[o] = ep1 * acc_s[r];
  }
  __syncthreads();
  if (tid < 16) {
    float sc = sqrtf(curv[0]);
    float nrm = fmaxf(sqrtf(rowsq[tid]), 1e-10f);
    float x = sc * nrm;
    esc[(size_t)h * Nn + row0 + tid] = tanhf(x) / x;   // th = esc * t
  }
}

// ---------- attention logits: att = -hyp_dist(th_i, th_j)^2 ; segmax via atomicMax ----------
__global__ void __launch_bounds__(256) k_att(
    const float* __restrict__ tb, const float* __restrict__ esc,
    const int* __restrict__ erow, const int* __restrict__ ecol,
    const float* __restrict__ curv, float* __restrict__ attb, unsigned* __restrict__ segmax) {
  int wid = (blockIdx.x * 256 + threadIdx.x) >> 5;
  if (wid >= Hc * Ee) return;
  const int lane = threadIdx.x & 31;
  const int h = wid / Ee, e = wid - h * Ee;
  const int i = erow[e], j = ecol[e];
  const float c = curv[0], sc = sqrtf(c);
  const float si = esc[(size_t)h * Nn + i], sj = esc[(size_t)h * Nn + j];
  const v4f ti = *(const v4f*)&tb[((size_t)h * Nn + i) * DOUTc + lane * 4];
  const v4f tj = *(const v4f*)&tb[((size_t)h * Nn + j) * DOUTc + lane * 4];
  const v4f u = ti * (-si);      // u = -th_i
  const v4f v = tj * sj;         // v = th_j
  float uv = 0.f, uu = 0.f, vv = 0.f;
#pragma unroll
  for (int q = 0; q < 4; ++q) { uv += u[q] * v[q]; uu += u[q] * u[q]; vv += v[q] * v[q]; }
  uv = wred(uv); uu = wred(uu); vv = wred(vv);
  const float A = 1.f + 2.f * c * uv + c * vv;
  const float B = 1.f - c * uu;
  const float den = fmaxf(1.f + 2.f * c * uv + c * c * uu * vv, 1e-10f);
  float s = 0.f;
#pragma unroll
  for (int q = 0; q < 4; ++q) { float ma = (A * u[q] + B * v[q]) / den; s += ma * ma; }
  s = wred(s);
  if (lane == 0) {
    float x = fminf(fmaxf(sc * sqrtf(s), 0.f), MAX_TANHf);
    float dist = (2.f / sc) * atanhf(x);
    float att = -dist * dist;
    attb[(size_t)h * Ee + e] = att;
    atomicMax(&segmax[(size_t)h * Nn + i], fmap(att));
  }
}

// ---------- exp(att - segmax) ; segment sum ----------
__global__ void k_exp(float* __restrict__ attb, const int* __restrict__ erow,
                      const unsigned* __restrict__ segmax, float* __restrict__ segsum) {
  int idx = blockIdx.x * 256 + threadIdx.x;
  if (idx >= Hc * Ee) return;
  int h = idx / Ee, e = idx - h * Ee;
  int i = erow[e];
  float m = funmap(segmax[(size_t)h * Nn + i]);
  float ev = __expf(attb[idx] - m);
  attb[idx] = ev;                       // in-place: attb now holds e-values
  atomicAdd(&segsum[(size_t)h * Nn + i], ev);
}

// ---------- msg: out[h][row] += alpha * t[h][col]; wave = 32-edge chunk, run-compressed atomics ----------
__global__ void __launch_bounds__(256) k_msg(
    const float* __restrict__ tb, const float* __restrict__ attb, const float* __restrict__ segsum,
    const int* __restrict__ erow, const int* __restrict__ ecol, float* __restrict__ outb) {
  const int nch = (Ee + 31) / 32;
  int wid = (blockIdx.x * 256 + threadIdx.x) >> 5;
  if (wid >= Hc * nch) return;
  const int lane = threadIdx.x & 31;
  const int h = wid / nch, ch = wid - h * nch;
  const int e0 = ch * 32, d0 = lane * 4;
  v4f acc = {0.f, 0.f, 0.f, 0.f};
  int cur = -1;
  for (int k = 0; k < 32; ++k) {
    int e = e0 + k;
    if (e >= Ee) break;
    int r = erow[e];
    if (r != cur) {                     // edge_row sorted -> long runs, few flushes
      if (cur >= 0) {
        float* dst = &outb[((size_t)h * Nn + cur) * DOUTc + d0];
#pragma unroll
        for (int q = 0; q < 4; ++q) atomicAdd(&dst[q], acc[q]);
      }
      cur = r; acc = (v4f){0.f, 0.f, 0.f, 0.f};
    }
    int cl = ecol[e];
    if (e + 1 < Ee)
      __builtin_prefetch(&tb[((size_t)h * Nn + ecol[e + 1]) * DOUTc + d0], 0, 1);
    float a = attb[(size_t)h * Ee + e] / segsum[(size_t)h * Nn + r];
    const v4f sv = *(const v4f*)&tb[((size_t)h * Nn + cl) * DOUTc + d0];
    acc += a * sv;
  }
  if (cur >= 0) {
    float* dst = &outb[((size_t)h * Nn + cur) * DOUTc + d0];
#pragma unroll
    for (int q = 0; q < 4; ++q) atomicAdd(&dst[q], acc[q]);
  }
}

// ---------- BN1 stats over nodes ----------
__global__ void k_bn1stats(const float* __restrict__ outb, float* __restrict__ s, float* __restrict__ q) {
  const int h = blockIdx.y, r0 = blockIdx.x * 250, tid = threadIdx.x;
  const int d = tid & 127, rr = tid >> 7;
  float su = 0.f, sq = 0.f;
  for (int r = rr; r < 250; r += 2) {
    float v = outb[((size_t)h * Nn + r0 + r) * DOUTc + d];
    su += v; sq += v * v;
  }
  atomicAdd(&s[h * DOUTc + d], su);
  atomicAdd(&q[h * DOUTc + d], sq);
}
__global__ void k_finstats(const float* s, const float* q, float* mu, float* rs, int n) {
  int i = blockIdx.x * 256 + threadIdx.x;
  if (i < n) {
    float m = s[i] / (float)Nn;
    float var = q[i] / (float)Nn - m * m;
    mu[i] = m; rs[i] = rsqrtf(var + BN_EPSf);
  }
}

// ---------- MLP layer 1: h = leaky_relu(BN1(out) @ w1 + b1), collect BN2 stats ----------
__global__ void __launch_bounds__(256) k_mlp1(
    const float* __restrict__ outb, const bf16_t* __restrict__ w1T,
    const float* __restrict__ b1, const float* __restrict__ g1, const float* __restrict__ bt1,
    const float* __restrict__ mu1, const float* __restrict__ rs1,
    bf16_t* __restrict__ hbuf, float* __restrict__ bn2s, float* __restrict__ bn2q) {
  __shared__ bf16_t atile[16 * HIDc];
  __shared__ float bsum[HIDc], bsq[HIDc];
  const int h = blockIdx.y, row0 = blockIdx.x * 16, tid = threadIdx.x;
  if (tid < HIDc) { bsum[tid] = 0.f; bsq[tid] = 0.f; }
#pragma unroll
  for (int p = 0; p < 8; ++p) {          // stage BN1(out) tile as bf16 A operand
    int idx = tid * 8 + p; int r = idx >> 7; int k = idx & 127;
    float v = outb[((size_t)h * Nn + row0 + r) * DOUTc + k];
    float ob = (v - mu1[h * DOUTc + k]) * rs1[h * DOUTc + k] * g1[k] + bt1[k];
    atile[idx] = f2bf(ob);
  }
  __syncthreads();
  const int lane = tid & 31, w = tid >> 5, lh = lane >> 4, m = lane & 15, n0 = w * 16;
  v8f acc = {};
#pragma unroll
  for (int kk = 0; kk < DOUTc / 32; ++kk) {
    v16bf a = ldfrag(&atile[m * DOUTc + kk * 32 + lh * 8]);
    v16bf b = ldfrag(&w1T[(size_t)(n0 + m) * DOUTc + kk * 32 + lh * 8]);
    acc = __builtin_amdgcn_wmma_f32_16x16x32_bf16(false, a, false, b, (short)0, acc, false, false);
  }
  const int n = n0 + m;
#pragma unroll
  for (int r = 0; r < 8; ++r) {
    float val = acc[r] + b1[n];
    val = val > 0.f ? val : 0.1f * val;  // leaky_relu(0.1)
    hbuf[((size_t)h * Nn + row0 + r + lh * 8) * HIDc + n] = f2bf(val);
    atomicAdd(&bsum[n], val);
    atomicAdd(&bsq[n], val * val);
  }
  __syncthreads();
  if (tid < HIDc) {                      // block-reduced BN2 partials -> global
    atomicAdd(&bn2s[h * HIDc + tid], bsum[tid]);
    atomicAdd(&bn2q[h * HIDc + tid], bsq[tid]);
  }
}

// ---------- MLP layer 2 + head mean: final = mean_h elu(BN2(h) @ w2 + b2) ----------
__global__ void __launch_bounds__(256) k_mlp2(
    const bf16_t* __restrict__ hbuf, const bf16_t* __restrict__ w2T,
    const float* __restrict__ b2, const float* __restrict__ g2, const float* __restrict__ bt2,
    const float* __restrict__ mu2, const float* __restrict__ rs2,
    const float* __restrict__ curv, float* __restrict__ dout) {
  __shared__ bf16_t atile[16 * HIDc];
  const int row0 = blockIdx.x * 16, tid = threadIdx.x;
  const int lane = tid & 31, w = tid >> 5, lh = lane >> 4, m = lane & 15, n0 = w * 16;
  v8f favg = {};
  for (int h = 0; h < Hc; ++h) {
    __syncthreads();                     // protect previous iteration's reads
#pragma unroll
    for (int p = 0; p < 8; ++p) {
      int idx = tid * 8 + p; int r = idx >> 7; int k = idx & 127;
      float v = bf2f(hbuf[((size_t)h * Nn + row0 + r) * HIDc + k]);
      float hb = (v - mu2[h * HIDc + k]) * rs2[h * HIDc + k] * g2[k] + bt2[k];
      atile[idx] = f2bf(hb);
    }
    __syncthreads();
    v8f acc = {};
#pragma unroll
    for (int kk = 0; kk < HIDc / 32; ++kk) {
      v16bf a = ldfrag(&atile[m * HIDc + kk * 32 + lh * 8]);
      v16bf b = ldfrag(&w2T[(size_t)(n0 + m) * HIDc + kk * 32 + lh * 8]);
      acc = __builtin_amdgcn_wmma_f32_16x16x32_bf16(false, a, false, b, (short)0, acc, false, false);
    }
#pragma unroll
    for (int r = 0; r < 8; ++r) {
      float val = acc[r] + b2[n0 + m];
      val = val > 0.f ? val : (__expf(val) - 1.f);   // elu
      favg[r] += val;
    }
  }
#pragma unroll
  for (int r = 0; r < 8; ++r)
    dout[(size_t)(row0 + r + lh * 8) * DOUTc + n0 + m] = favg[r] * (1.f / Hc);
  if (blockIdx.x == 0 && tid == 0) dout[(size_t)Nn * DOUTc] = curv[0];  // tuple: (final, curvature)
}

// ---------- host ----------
extern "C" void kernel_launch(void* const* d_in, const int* in_sizes, int n_in,
                              void* d_out, int out_size, void* d_ws, size_t ws_size,
                              hipStream_t stream) {
  const float* x    = (const float*)d_in[0];
  const int*   erow = (const int*)d_in[1];
  const int*   ecol = (const int*)d_in[2];
  const float* Wt   = (const float*)d_in[3];
  const float* Ws   = (const float*)d_in[4];
  const float* w1   = (const float*)d_in[5];
  const float* b1   = (const float*)d_in[6];
  const float* w2   = (const float*)d_in[7];
  const float* b2   = (const float*)d_in[8];
  const float* g1   = (const float*)d_in[9];
  const float* bt1  = (const float*)d_in[10];
  const float* g2   = (const float*)d_in[11];
  const float* bt2  = (const float*)d_in[12];
  const float* curv = (const float*)d_in[13];
  const float* eps  = (const float*)d_in[14];
  float* fout = (float*)d_out;
  (void)in_sizes; (void)n_in; (void)out_size; (void)ws_size;

  char* ws = (char*)d_ws; size_t off = 0;
  auto alloc = [&](size_t bytes) -> char* {
    char* p = ws + off; off += (bytes + 255) & ~(size_t)255; return p;
  };
  bf16_t* xb   = (bf16_t*)alloc((size_t)Nn * DINc * 2);
  bf16_t* wtT  = (bf16_t*)alloc((size_t)Hc * DINc * DOUTc * 2);
  bf16_t* wsT  = (bf16_t*)alloc((size_t)Hc * DINc * DOUTc * 2);
  bf16_t* w1T  = (bf16_t*)alloc((size_t)DOUTc * HIDc * 2);
  bf16_t* w2T  = (bf16_t*)alloc((size_t)HIDc * DOUTc * 2);
  float*  tb   = (float*)alloc((size_t)Hc * Nn * DOUTc * 4);
  float*  outb = (float*)alloc((size_t)Hc * Nn * DOUTc * 4);
  bf16_t* hbuf = (bf16_t*)alloc((size_t)Hc * Nn * HIDc * 2);
  float*  esc  = (float*)alloc((size_t)Hc * Nn * 4);
  float*  attb = (float*)alloc((size_t)Hc * Ee * 4);
  // contiguous accumulator region (all sizes are multiples of 256B -> no padding gaps)
  unsigned* segmax = (unsigned*)alloc((size_t)Hc * Nn * 4);
  float*  segsum = (float*)alloc((size_t)Hc * Nn * 4);
  float*  bn1s = (float*)alloc((size_t)Hc * DOUTc * 4);
  float*  bn1q = (float*)alloc((size_t)Hc * DOUTc * 4);
  float*  bn2s = (float*)alloc((size_t)Hc * HIDc * 4);
  float*  bn2q = (float*)alloc((size_t)Hc * HIDc * 4);
  float*  mu1  = (float*)alloc((size_t)Hc * DOUTc * 4);
  float*  rs1  = (float*)alloc((size_t)Hc * DOUTc * 4);
  float*  mu2  = (float*)alloc((size_t)Hc * HIDc * 4);
  float*  rs2  = (float*)alloc((size_t)Hc * HIDc * 4);

  // 0) zero accumulators (segmax init 0u == mapped -inf; graph-replay safe)
  int zn = 2 * Hc * Nn + 2 * Hc * DOUTc + 2 * Hc * HIDc;
  k_zero<<<(zn + 255) / 256, 256, 0, stream>>>((float*)segmax, zn);
  // 1) precision conversion / operand layout
  k_f2bf<<<(Nn * DINc + 255) / 256, 256, 0, stream>>>(x, xb, Nn * DINc);
  dim3 gtw((DINc * DOUTc + 255) / 256, Hc);
  k_tr<<<gtw, 256, 0, stream>>>(Wt, wtT, DINc, DOUTc);
  k_tr<<<gtw, 256, 0, stream>>>(Ws, wsT, DINc, DOUTc);
  k_tr<<<dim3((DOUTc * HIDc + 255) / 256, 1), 256, 0, stream>>>(w1, w1T, DOUTc, HIDc);
  k_tr<<<dim3((HIDc * DOUTc + 255) / 256, 1), 256, 0, stream>>>(w2, w2T, HIDc, DOUTc);
  // 2) fused dual-WMMA projections + expmap0 scale + out init
  k_gemm_proj<<<dim3(Nn / 16, Hc), 256, 0, stream>>>(xb, wtT, wsT, curv, eps, tb, outb, esc);
  // 3) attention logits + segment max (wave per head-edge)
  k_att<<<(Hc * Ee) / 8, 256, 0, stream>>>(tb, esc, erow, ecol, curv, attb, segmax);
  // 4) exp + segment sum
  k_exp<<<(Hc * Ee + 255) / 256, 256, 0, stream>>>(attb, erow, segmax, segsum);
  // 5) messages (run-compressed scatter-add over sorted rows)
  const int nch = (Ee + 31) / 32;
  k_msg<<<(Hc * nch + 7) / 8, 256, 0, stream>>>(tb, attb, segsum, erow, ecol, outb);
  // 6) BN1 stats -> mu/rsig
  k_bn1stats<<<dim3(40, Hc), 256, 0, stream>>>(outb, bn1s, bn1q);
  k_finstats<<<(Hc * DOUTc + 255) / 256, 256, 0, stream>>>(bn1s, bn1q, mu1, rs1, Hc * DOUTc);
  // 7) MLP layer 1 (WMMA) + BN2 stats
  k_mlp1<<<dim3(Nn / 16, Hc), 256, 0, stream>>>(outb, w1T, b1, g1, bt1, mu1, rs1, hbuf, bn2s, bn2q);
  k_finstats<<<(Hc * HIDc + 255) / 256, 256, 0, stream>>>(bn2s, bn2q, mu2, rs2, Hc * HIDc);
  // 8) MLP layer 2 (WMMA) + head mean + curvature passthrough
  k_mlp2<<<Nn / 16, 256, 0, stream>>>(hbuf, w2T, b2, g2, bt2, mu2, rs2, curv, fout);
}
